// QGNNTrafficPredictor_36240934043943
// MI455X (gfx1250) — matrix operand order repaired and checked by reading, not compiled, not run on hardware
//
#include <hip/hip_runtime.h>
#include <hip/hip_bf16.h>

typedef __attribute__((ext_vector_type(16))) _Float16 v16h;
typedef __attribute__((ext_vector_type(8)))  _Float16 v8h;
typedef __attribute__((ext_vector_type(8)))  float    v8f;

#define BB 128
#define LL 12
#define NN 307
#define NP 320
#define CC 4
#define HIDN 64
#define GATES 192
#define FIN 75
#define FKP 96
#define NROWS (BB*NN)      /* 39296 */
#define NTILES (NROWS/16)  /* 2456  */
#define BLC (BB*LL)        /* 1536  */

// ---- workspace byte offsets (all 256-aligned) ----
#define WS_AN    ((size_t)0)                        // f16 [NP*NP] row-major (K-contig)
#define WS_HAM1T ((size_t)204800)                   // f16 [64*64] transposed (N-major,K-contig)
#define WS_HAM0  ((size_t)212992)                   // f32 [4*64]
#define WS_BNA   ((size_t)214016)                   // f32 [2*64]
#define WS_BNB   ((size_t)214528)                   // f32 [2*64]
#define WS_DINV  ((size_t)215040)                   // f32 [NN]
#define WS_WIHT  ((size_t)216320)                   // f16 [192*96] transposed, K-padded
#define WS_WHHT  ((size_t)253184)                   // f16 [192*64] transposed
#define WS_S     ((size_t)277760)                   // f16 [BLC*64*NP] TRANSPOSED support
#define WS_X     ((size_t)(277760 + 62914560))      // f16 [BLC*NP*64] row-major
#define WS_FEAT  ((size_t)(277760 + 125829120))     // f16 [NROWS*LL*96] K-padded

// ---------- WMMA fragment loaders: fully vectorized (b128) ----------
// A: 16x32 f16, row-major K-contiguous, leading dim lda (elements).
// Lane layout: elements 0..15 = K hlf*16..hlf*16+15 contiguous -> one 32B load.
__device__ __forceinline__ v16h load_A_frag(const _Float16* A, int lda) {
  int lane = threadIdx.x & 31;
  int hlf  = lane >> 4;
  int m    = lane & 15;
  const _Float16* p = A + (size_t)m*lda + hlf*16;
  return *(const v16h*)(p);
}

// B: 32x16 f16 stored TRANSPOSED (N-major, K-contiguous), leading dim ldk.
// Lane layout: elements 0..15 = K hlf*16..hlf*16+15 contiguous -> one 32B load.
__device__ __forceinline__ v16h load_B_fragT(const _Float16* BT, int ldk) {
  int lane = threadIdx.x & 31;
  int hlf  = lane >> 4;
  int n    = lane & 15;
  return *(const v16h*)(BT + (size_t)n*ldk + hlf*16);
}

#define WMMA(a,b,c) __builtin_amdgcn_wmma_f32_16x16x32_f16(false, (a), false, (b), (short)0, (c), false, false)

// ---------- small prep: hamilton matrices + folded BN ----------
__global__ void k_small_prep(const float* w0, const float* w1,
                             const float* g, const float* be, const float* mu, const float* va,
                             float* ham0, _Float16* ham1T, float* bna, float* bnb) {
  int idx = blockIdx.x*blockDim.x + threadIdx.x;
  const int   cmp[4][4] = {{0,1,2,3},{1,0,3,2},{2,3,0,1},{3,2,1,0}};
  const float sgn[4][4] = {{1.f,-1.f,-1.f,-1.f},{1.f,1.f,-1.f,1.f},{1.f,1.f,1.f,-1.f},{1.f,-1.f,1.f,1.f}};
  if (idx < 64*64) {                  // ham1 element (rr=k, cc=n) -> store transposed [n][k]
    int rr = idx >> 6, cc = idx & 63;
    int rb = rr >> 4, p = rr & 15, cb = cc >> 4, q = cc & 15;
    float v = sgn[cb][rb] * w1[p*64 + cmp[cb][rb]*16 + q];
    ham1T[cc*64 + rr] = (_Float16)v;
  } else if (idx < 64*64 + 256) {     // ham0: (4,64) row-major f32
    int j = idx - 64*64;
    int rb = j >> 6, cc = j & 63;
    int cb = cc >> 4, q = cc & 15;
    ham0[rb*64 + cc] = sgn[cb][rb] * w0[cmp[cb][rb]*16 + q];
  } else if (idx < 64*64 + 256 + 128) { // BN affine fold
    int j = idx - (64*64 + 256);
    float s = g[j] * rsqrtf(va[j] + 1e-5f);
    bna[j] = s;
    bnb[j] = be[j] - mu[j]*s;
  }
}

// ---------- GRU weight transpose+convert (done once) ----------
__global__ void k_wprep(const float* w_ih, const float* w_hh, _Float16* WihT, _Float16* WhhT) {
  int idx = blockIdx.x*blockDim.x + threadIdx.x;
  if (idx < GATES*FKP) {
    int nn = idx / FKP, k = idx % FKP;
    WihT[idx] = (k < FIN) ? (_Float16)w_ih[nn*FIN + k] : (_Float16)0.f;
  } else if (idx < GATES*FKP + GATES*64) {
    int j = idx - GATES*FKP;
    int nn = j / 64, k = j % 64;
    WhhT[j] = (_Float16)w_hh[nn*64 + k];
  }
}

__global__ void k_dinv(const float* adj, float* dinv) {
  int n = blockIdx.x*blockDim.x + threadIdx.x;
  if (n < NN) {
    float s = 1.0f;                   // self loop
    for (int m = 0; m < NN; ++m) s += adj[n*NN + m];
    dinv[n] = rsqrtf(s);
  }
}

__global__ void k_an(const float* adj, const float* dinv, _Float16* an) {
  int idx = blockIdx.x*blockDim.x + threadIdx.x;
  if (idx >= NP*NP) return;
  int n = idx / NP, m = idx % NP;
  float v = 0.f;
  if (n < NN && m < NN) {
    float a = adj[n*NN + m] + (n == m ? 1.f : 0.f);
    v = dinv[n]*a*dinv[m];
  }
  an[idx] = (_Float16)v;
}

// ---------- support0: (B,L,N,4) x ham0(4,64) -> TRANSPOSED f16 ST[bl][d][m] ----------
__global__ void k_support0(const float* hist, const float* ham0, _Float16* ST) {
  int bl = blockIdx.x >> 6;
  int d  = blockIdx.x & 63;
  int m  = threadIdx.x;          // 320 threads, coalesced along m
  float v = 0.f;
  if (m < NN) {
    const float* x = &hist[(size_t)(bl*NN + m)*CC];
#pragma unroll
    for (int c = 0; c < 4; ++c) v += x[c]*ham0[c*64 + d];
  }
  ST[((size_t)bl*HIDN + d)*NP + m] = (_Float16)v;
}

// ---------- graph aggregation GEMM + BN + tanh : LDS-free WMMA ----------
__global__ void k_graphconv(const _Float16* an, const _Float16* ST,
                            const float* bna, const float* bnb,
                            _Float16* Xout, _Float16* featOut, int layer) {
  int bl   = blockIdx.x;
  int row0 = blockIdx.y * 16;
  int wave = threadIdx.x >> 5;
  int lane = threadIdx.x & 31;
  int hlf  = lane >> 4;
  int nl   = lane & 15;
  int col0 = wave * 16;
  const _Float16* Abase = &an[(size_t)row0 * NP];
  const _Float16* Bbase = &ST[((size_t)bl*HIDN + col0) * NP];
  v8f acc = {};
#pragma unroll
  for (int k0 = 0; k0 < NP; k0 += 32) {
    v16h a = load_A_frag(Abase + k0, NP);
    v16h b = load_B_fragT(Bbase + k0, NP);
    acc = WMMA(a, b, acc);
  }
  int d = col0 + nl;
  float sa = bna[layer*64 + d], sb = bnb[layer*64 + d];
#pragma unroll
  for (int r = 0; r < 8; ++r) {
    int m = row0 + r + hlf*8;
    float v = tanhf(acc[r]*sa + sb);
    if (layer == 0) {       // X row-major [bl][m][64], zero pad rows for next GEMM
      Xout[((size_t)bl*NP + m)*HIDN + d] = (m < NN) ? (_Float16)v : (_Float16)0.f;
    } else if (m < NN) {    // feat [row][l][96]
      int b_ = bl / LL, l = bl % LL;
      size_t row = (size_t)b_*NN + m;
      featOut[(row*LL + l)*FKP + d] = (_Float16)v;
    }
  }
}

// ---------- support1: X[m][64] x ham1 -> TRANSPOSED ST[bl][d][m] (WMMA, LDS-free) ----------
__global__ void k_support1(const _Float16* X, const _Float16* ham1T, _Float16* ST) {
  int bl   = blockIdx.x;
  int row0 = blockIdx.y * 16;
  int wave = threadIdx.x >> 5;
  int lane = threadIdx.x & 31;
  int hlf  = lane >> 4;
  int nl   = lane & 15;
  int col0 = wave * 16;
  const _Float16* Abase = &X[((size_t)bl*NP + row0) * HIDN];
  v8f acc = {};
#pragma unroll
  for (int k0 = 0; k0 < 64; k0 += 32) {
    v16h a = load_A_frag(Abase + k0, HIDN);
    v16h b = load_B_fragT(&ham1T[(size_t)col0*64 + k0], 64);
    acc = WMMA(a, b, acc);
  }
#pragma unroll
  for (int r = 0; r < 8; ++r) {
    int m = row0 + r + hlf*8;   // pad rows of X are zero -> acc zero -> clean K-pad
    ST[((size_t)bl*HIDN + col0 + nl)*NP + m] = (_Float16)acc[r];
  }
}

// ---------- feat tail: speed + time/day embeddings, zero K-pad ----------
__global__ void k_featfill(const float* hist, const float* tid_emb, const float* dw_emb, _Float16* feat) {
  int idx = blockIdx.x*blockDim.x + threadIdx.x;
  if (idx >= BB*NN*LL) return;
  int l = idx % LL; int rn = idx / LL;
  int b = rn / NN; int n = rn % NN;
  const float* h = &hist[((size_t)(b*LL + l)*NN + n)*CC];
  int tix = (int)roundf(h[1]*287.f); tix = tix < 0 ? 0 : (tix > 287 ? 287 : tix);
  int dix = (int)roundf(h[2]*6.f);   dix = dix < 0 ? 0 : (dix > 6   ? 6   : dix);
  _Float16* f = &feat[((size_t)rn*LL + l)*FKP];
  f[64] = (_Float16)h[0];
#pragma unroll
  for (int j = 0; j < 5; ++j) f[65+j] = (_Float16)tid_emb[tix*5 + j];
#pragma unroll
  for (int j = 0; j < 5; ++j) f[70+j] = (_Float16)dw_emb[dix*5 + j];
#pragma unroll
  for (int j = 75; j < 96; ++j) f[j] = (_Float16)0.f;
}

// ---------- persistent GRU: gates in registers, f32 recurrence in v8f ----------
// Wave w owns hidden units d in [16w,16w+16) for ALL three gates (cols 16w, 64+16w, 128+16w).
__global__ void k_gru(const _Float16* feat, const _Float16* WihT, const _Float16* WhhT,
                      const float* b_ih, const float* b_hh,
                      const float* fc_w, const float* fc_b, float* out) {
  __shared__ __align__(32) _Float16 htile[16*64];
  int tid  = threadIdx.x;
  int wave = tid >> 5, lane = tid & 31, hlf = lane >> 4, nl = lane & 15;
  size_t rowbase = (size_t)blockIdx.x * 16;
  int d = wave*16 + nl;                 // this lane's hidden unit
  float bir = b_ih[d], biz = b_ih[64+d], bin = b_ih[128+d];
  float bhr = b_hh[d], bhz = b_hh[64+d], bhn = b_hh[128+d];
  const _Float16* WihR = &WihT[(size_t)(d - nl)*FKP];          // col0 = wave*16
  const _Float16* WihZ = &WihT[(size_t)(64 + d - nl)*FKP];
  const _Float16* WihN = &WihT[(size_t)(128 + d - nl)*FKP];
  const _Float16* WhhR = &WhhT[(size_t)(d - nl)*64];
  const _Float16* WhhZ = &WhhT[(size_t)(64 + d - nl)*64];
  const _Float16* WhhN = &WhhT[(size_t)(128 + d - nl)*64];

  for (int i = tid; i < 16*64; i += 128) htile[i] = (_Float16)0.f;
  v8f hreg = {};                        // f32 hidden state for this lane's 8 (m,d) cells
  __syncthreads();

  for (int t = 0; t < LL; ++t) {
    const _Float16* featbase = &feat[(rowbase*LL + t)*FKP];    // A[m][k], lda = LL*FKP
    v8f air = {}, aiz = {}, ain = {};
#pragma unroll
    for (int k0 = 0; k0 < FKP; k0 += 32) {
      v16h a = load_A_frag(featbase + k0, LL*FKP);
      air = WMMA(a, load_B_fragT(WihR + k0, FKP), air);
      aiz = WMMA(a, load_B_fragT(WihZ + k0, FKP), aiz);
      ain = WMMA(a, load_B_fragT(WihN + k0, FKP), ain);
    }
    v8f ahr = {}, ahz = {}, ahn = {};
#pragma unroll
    for (int k0 = 0; k0 < 64; k0 += 32) {
      v16h a = load_A_frag(&htile[k0], 64);
      ahr = WMMA(a, load_B_fragT(WhhR + k0, 64), ahr);
      ahz = WMMA(a, load_B_fragT(WhhZ + k0, 64), ahz);
      ahn = WMMA(a, load_B_fragT(WhhN + k0, 64), ahn);
    }
    __syncthreads();                    // all waves done reading htile
#pragma unroll
    for (int r = 0; r < 8; ++r) {
      float rg = 1.f/(1.f + expf(-(air[r] + bir + ahr[r] + bhr)));
      float zg = 1.f/(1.f + expf(-(aiz[r] + biz + ahz[r] + bhz)));
      float nc = tanhf(ain[r] + bin + rg*(ahn[r] + bhn));
      float hnew = (1.f - zg)*nc + zg*hreg[r];
      hreg[r] = hnew;
      int m = r + hlf*8;
      htile[m*64 + d] = (_Float16)hnew;
    }
    __syncthreads();
    // last PRE steps: relu + fc -> output (B,3,N,1)
    if (t >= LL - 3 && tid < 16) {
      size_t row = rowbase + tid;
      float s = fc_b[0];
#pragma unroll
      for (int dd = 0; dd < 64; ++dd) {
        float h = (float)htile[tid*64 + dd];
        s += (h > 0.f ? h : 0.f) * fc_w[dd];
      }
      int b_ = (int)(row / NN), n = (int)(row % NN);
      out[((size_t)b_*3 + (t - (LL - 3)))*NN + n] = s;
    }
  }
}

extern "C" void kernel_launch(void* const* d_in, const int* in_sizes, int n_in,
                              void* d_out, int out_size, void* d_ws, size_t ws_size,
                              hipStream_t stream) {
  const float* hist    = (const float*)d_in[0];
  const float* adj     = (const float*)d_in[1];
  const float* w0      = (const float*)d_in[2];
  const float* w1      = (const float*)d_in[3];
  const float* bn_g    = (const float*)d_in[4];
  const float* bn_be   = (const float*)d_in[5];
  const float* bn_mu   = (const float*)d_in[6];
  const float* bn_va   = (const float*)d_in[7];
  const float* tid_emb = (const float*)d_in[8];
  const float* dw_emb  = (const float*)d_in[9];
  const float* w_ih    = (const float*)d_in[10];
  const float* w_hh    = (const float*)d_in[11];
  const float* b_ih    = (const float*)d_in[12];
  const float* b_hh    = (const float*)d_in[13];
  const float* fc_w    = (const float*)d_in[14];
  const float* fc_b    = (const float*)d_in[15];

  char* ws = (char*)d_ws;
  _Float16* an    = (_Float16*)(ws + WS_AN);
  _Float16* ham1T = (_Float16*)(ws + WS_HAM1T);
  float*    ham0  = (float*)   (ws + WS_HAM0);
  float*    bna   = (float*)   (ws + WS_BNA);
  float*    bnb   = (float*)   (ws + WS_BNB);
  float*    dinv  = (float*)   (ws + WS_DINV);
  _Float16* WihT  = (_Float16*)(ws + WS_WIHT);
  _Float16* WhhT  = (_Float16*)(ws + WS_WHHT);
  _Float16* Sbuf  = (_Float16*)(ws + WS_S);
  _Float16* Xbuf  = (_Float16*)(ws + WS_X);
  _Float16* feat  = (_Float16*)(ws + WS_FEAT);
  float*    out   = (float*)d_out;

  k_small_prep<<<18, 256, 0, stream>>>(w0, w1, bn_g, bn_be, bn_mu, bn_va, ham0, ham1T, bna, bnb);
  k_wprep<<<120, 256, 0, stream>>>(w_ih, w_hh, WihT, WhhT);
  k_dinv<<<2, 256, 0, stream>>>(adj, dinv);
  k_an<<<(NP*NP + 255)/256, 256, 0, stream>>>(adj, dinv, an);
  k_support0<<<BLC*HIDN, NP, 0, stream>>>(hist, ham0, Sbuf);
  k_graphconv<<<dim3(BLC, 20), 128, 0, stream>>>(an, Sbuf, bna, bnb, Xbuf, (_Float16*)nullptr, 0);
  k_support1<<<dim3(BLC, 20), 128, 0, stream>>>(Xbuf, ham1T, Sbuf);
  k_graphconv<<<dim3(BLC, 20), 128, 0, stream>>>(an, Sbuf, bna, bnb, (_Float16*)nullptr, feat, 1);
  k_featfill<<<(BB*NN*LL + 255)/256, 256, 0, stream>>>(hist, tid_emb, dw_emb, feat);
  k_gru<<<NTILES, 128, 0, stream>>>(feat, WihT, WhhT, b_ih, b_hh, fc_w, fc_b, out);
}
